// LSTMSOH_58841051955297
// MI455X (gfx1250) — matrix operand order repaired and checked by reading, not compiled, not run on hardware
//
#include <hip/hip_runtime.h>

#define INPUT_DIM 4
#define HIDDEN    80
#define LAYERS    5
#define PRED_LEN  48
#define BATCH     512
#define SEQ       336
#define G         320     // 4*HIDDEN
#define NTILES    20      // G/16
#define KP        96      // HIDDEN padded to 3 chunks of 32
#define BT        16      // batch tile (WMMA M)
#define NW        5       // waves per block: wave w owns hidden slice [16w,16w+16)
#define NTHREADS  (NW*32) // 160

// dynamic-LDS weight cache: Wih0 (20 frags) + Whh layers 0..3 (240 frags)
#define C_WIH0    0
#define C_WHH     20
#define C_FRAGS   260                 // 260 KB

typedef __attribute__((ext_vector_type(16))) __bf16 v16bf;
typedef __attribute__((ext_vector_type(8)))  float  v8f;

union FragU { unsigned int u[8]; v16bf v; };

__device__ __forceinline__ unsigned short f2bf(float x) {
  unsigned int u = __float_as_uint(x);
  u += 0x7FFFu + ((u >> 16) & 1u);   // round-to-nearest-even
  return (unsigned short)(u >> 16);
}
// branch-free fast transcendentals (v_exp_f32 + v_rcp_f32)
__device__ __forceinline__ float sig_(float x) {
  return __builtin_amdgcn_rcpf(1.0f + __expf(-x));
}
__device__ __forceinline__ float tanh_(float x) {
  float e = __expf(-2.0f * x);
  return (1.0f - e) * __builtin_amdgcn_rcpf(1.0f + e);
}
// WGP-scope prefetch: pulls into WGP$ + L2 (default scope 0)
__device__ __forceinline__ void prefetch_wgp(const void* p) {
  asm volatile("global_prefetch_b8 %0, off" :: "v"(p));
}

__device__ __forceinline__ void loadA(FragU& A, const unsigned short* row, int off) {
  // row = &buf[lr][0]; off = kc*32 + hi*8 (dword aligned)
  const unsigned int* pa = (const unsigned int*)(row + off);
  A.u[0] = pa[0]; A.u[1] = pa[1]; A.u[2] = pa[2];  A.u[3] = pa[3];
  A.u[4] = pa[8]; A.u[5] = pa[9]; A.u[6] = pa[10]; A.u[7] = pa[11];
}
// B fragment: per-lane 32 B at (base + compile-time byte offset); base already
// includes wave/lane components so the offset folds into the load immediate.
__device__ __forceinline__ void loadBo(FragU& B, const void* base, int byteOff) {
  const uint4* q = (const uint4*)((const char*)base + byteOff);
  uint4 q0 = q[0], q1 = q[1];
  B.u[0] = q0.x; B.u[1] = q0.y; B.u[2] = q0.z; B.u[3] = q0.w;
  B.u[4] = q1.x; B.u[5] = q1.y; B.u[6] = q1.z; B.u[7] = q1.w;
}
__device__ __forceinline__ v8f wmma_bf16(const FragU& A, const FragU& B, v8f c) {
  return __builtin_amdgcn_wmma_f32_16x16x32_bf16(false, A.v, false, B.v,
                                                 (short)0, c, false, false);
}

// ---------------------------------------------------------------------------
// Pre-pass: f32 weights -> bf16, pre-swizzled into WMMA B-fragment order.
// Fragment = 32 lanes x 16 bf16 (1 KB); lane holds column n = lane&15,
// K = (el&7) + 8*(lane>=16) + 16*(el>>3); B[k][n] = W[n][kc*32+K]; zero-pad kglobal >= Kdim.
// Fragment index f = layer*(20*kcN) + tile*kcN + kc.
// ---------------------------------------------------------------------------
__global__ void conv_weights(const float* __restrict__ W, int Kdim, int kcN,
                             unsigned short* __restrict__ dst) {
  int f = blockIdx.x;
  int perLayer = NTILES * kcN;
  int layer = f / perLayer;
  int rem   = f % perLayer;
  int tile  = rem / kcN;
  int kc    = rem % kcN;
  const float* Wl = W + (size_t)layer * G * Kdim;
  unsigned short* d = dst + (size_t)f * 512;
  for (int idx = threadIdx.x; idx < 512; idx += blockDim.x) {
    int lane = idx >> 4;
    int el   = idx & 15;
    int n    = tile * 16 + (lane & 15);
    int kg   = kc * 32 + (el & 7) + ((el >> 3) << 4) + ((lane >> 4) << 3);
    float v  = (kg < Kdim) ? Wl[(size_t)n * Kdim + kg] : 0.0f;
    d[idx] = f2bf(v);
  }
}

__global__ void conv_bias(const float* __restrict__ bih, const float* __restrict__ bhh,
                          float* __restrict__ dst, int n) {
  int i = blockIdx.x * blockDim.x + threadIdx.x;
  if (i < n) dst[i] = bih[i] + bhh[i];
}

// ---------------------------------------------------------------------------
// Main kernel.  One block owns 16 batch rows for the whole enc+dec sequence.
// 5 waves; wave w computes gate tiles {w, w+5, w+10, w+15} = (i,f,g,o) for
// hidden slice [16w,16w+16): 4 independent WMMA acc chains sharing each A
// fragment; chains start from inline-0 C and the bias is folded into the
// register-resident LSTM cell update (one barrier per layer).
// Wih0 + Whh layers 0..3 cached in dynamic LDS (260 KB); Whh layer 4 and
// Wih layers 1..4 stream from L2 with WGP-scope prefetch a layer ahead.
// ---------------------------------------------------------------------------
__global__ __launch_bounds__(NTHREADS)
void lstm_seq2seq(const float* __restrict__ x,
                  const float* __restrict__ ff,
                  const unsigned short* __restrict__ eWih0,
                  const unsigned short* __restrict__ eWih,
                  const unsigned short* __restrict__ eWhh,
                  const float* __restrict__ eBias,
                  const unsigned short* __restrict__ dWih0,
                  const unsigned short* __restrict__ dWih,
                  const unsigned short* __restrict__ dWhh,
                  const float* __restrict__ dBias,
                  const float* __restrict__ fcW,
                  const float* __restrict__ fcb,
                  float* __restrict__ out) {
  __shared__ float          sc[LAYERS][BT][HIDDEN];     // cell state, f32
  __shared__ unsigned short hbf[LAYERS][BT][KP];        // h state, bf16, K-padded
  __shared__ unsigned short inbf[BT][KP];               // current layer input, bf16
  __shared__ float          htop[BT][HIDDEN];           // top-layer h for FC
  __shared__ float          pred16[BT];
  __shared__ float          sfcW[HIDDEN];
  extern __shared__ unsigned short wC[];                // weight fragment cache

  const int tid  = threadIdx.x;
  const int lane = tid & 31;
  const int w    = tid >> 5;       // wave id == hidden slice (0..4)
  const int hi   = lane >> 4;
  const int lr   = lane & 15;
  const int b0   = blockIdx.x * BT;
  const int jj   = w * 16 + lr;    // this lane's hidden column

  for (int i = tid; i < LAYERS*BT*HIDDEN; i += NTHREADS) (&sc[0][0][0])[i]  = 0.0f;
  for (int i = tid; i < LAYERS*BT*KP;     i += NTHREADS) (&hbf[0][0][0])[i] = 0;
  for (int i = tid; i < BT*KP;            i += NTHREADS) (&inbf[0][0])[i]   = 0;
  for (int i = tid; i < HIDDEN;           i += NTHREADS) sfcW[i] = fcW[i];
  const float fcb0 = fcb[0];

  float bias20[LAYERS * 4];

  auto loadPhase = [&](const unsigned short* Wih0p, const unsigned short* Whhp,
                       const float* Biasp) {
    #pragma unroll
    for (int l = 0; l < LAYERS; ++l)
      #pragma unroll
      for (int g = 0; g < 4; ++g)
        bias20[l * 4 + g] = Biasp[l * G + g * HIDDEN + jj];
    // cooperative copy into LDS: Wih0 (20 frags) + Whh layers 0..3 (240 frags)
    {
      const uint4* s0 = (const uint4*)Wih0p;
      uint4* d0 = (uint4*)(wC + (size_t)C_WIH0 * 512);
      for (int i = tid; i < 20 * 512 / 8; i += NTHREADS) d0[i] = s0[i];
      const uint4* s1 = (const uint4*)Whhp;
      uint4* d1 = (uint4*)(wC + (size_t)C_WHH * 512);
      for (int i = tid; i < 240 * 512 / 8; i += NTHREADS) d1[i] = s1[i];
    }
  };
  loadPhase(eWih0, eWhh, eBias);

  // cell update: bias folded here (acc chains start from inline 0)
  auto cellUpdate = [&](int l, v8f gi, v8f gf, v8f gg, v8f go) {
    const float bi = bias20[l * 4 + 0], bfv = bias20[l * 4 + 1];
    const float bg = bias20[l * 4 + 2], bo  = bias20[l * 4 + 3];
    #pragma unroll
    for (int v = 0; v < 8; ++v) {
      int b = v + 8 * hi;
      float cn = sig_(gf[v] + bfv) * sc[l][b][jj] +
                 sig_(gi[v] + bi)  * tanh_(gg[v] + bg);
      float hn = sig_(go[v] + bo) * tanh_(cn);
      sc[l][b][jj] = cn;
      unsigned short hb = f2bf(hn);
      hbf[l][b][jj] = hb;   // recurrent input for next timestep
      inbf[b][jj]   = hb;   // feedforward input for next layer
      if (l == LAYERS - 1) htop[b][jj] = hn;
    }
  };

  // phase-invariant LDS fragment bases (include wave + lane components)
  const unsigned short* wC0B = wC + (size_t)(C_WIH0 + w) * 512 + lane * 16;
  const unsigned short* wChB = wC + (size_t)(C_WHH + w * 3) * 512 + lane * 16;
  __syncthreads();

  for (int t = 0; t < SEQ + PRED_LEN; ++t) {
    const bool enc = (t < SEQ);
    if (t == SEQ) loadPhase(dWih0, dWhh, dBias);   // switch to decoder weights

    // ---- stage layer-0 input (4 valid cols; padded B cols are zero, so
    //      stale values in inbf[.][4..] are annihilated by the GEMM) ----
    if (enc) {
      if (tid < BT * INPUT_DIM) {
        int b = tid >> 2, d2 = tid & 3;
        inbf[b][d2] = f2bf(x[((size_t)(b0 + b) * SEQ + t) * INPUT_DIM + d2]);
      }
    } else if (t == SEQ) {
      if (tid < BT * INPUT_DIM) {
        int b = tid >> 2, d2 = tid & 3;
        inbf[b][d2] = f2bf(x[((size_t)(b0 + b) * SEQ + (SEQ - 1)) * INPUT_DIM + d2]);
      }
    } else {
      int s = t - SEQ;   // >= 1
      if (tid < BT) {
        inbf[tid][0] = f2bf(pred16[tid]);
      } else if (tid < BT + BT * 3) {
        int q = tid - BT; int b = q / 3, d2 = q % 3;
        inbf[b][1 + d2] =
            f2bf(ff[((size_t)(b0 + b) * PRED_LEN + (s - 1)) * 3 + d2]);
      }
    }
    // per-phase global fragment bases (one address pair each)
    const unsigned short* WihB =
        (enc ? eWih : dWih) + (size_t)w * 3 * 512 + lane * 16;       // layers 1..4 ih
    const unsigned short* Whh4B =
        (enc ? eWhh : dWhh) + (size_t)(240 + w * 3) * 512 + lane * 16; // layer-4 hh
    __syncthreads();

    // ---------------- layer 0 (all weights from LDS cache) ----------------
    {
      v8f acc[4];
      FragU A, B;
      loadA(A, &inbf[lr][0], hi * 8);
      #pragma unroll
      for (int g = 0; g < 4; ++g) {
        loadBo(B, wC0B, g * 5 * 1024);
        v8f z{};
        acc[g] = wmma_bf16(A, B, z);            // C = inline 0
      }
      #pragma unroll
      for (int kc = 0; kc < 3; ++kc) {
        loadA(A, &hbf[0][lr][0], kc * 32 + hi * 8);
        #pragma unroll
        for (int g = 0; g < 4; ++g) {
          loadBo(B, wChB, (g * 15 + kc) * 1024);
          acc[g] = wmma_bf16(A, B, acc[g]);
        }
      }
      // prefetch layer-1 Wih fragments into WGP$ while updating
      #pragma unroll
      for (int g = 0; g < 4; ++g)
        #pragma unroll
        for (int kc = 0; kc < 3; ++kc)
          prefetch_wgp((const char*)WihB + (g * 15 + kc) * 1024);
      cellUpdate(0, acc[0], acc[1], acc[2], acc[3]);
      __syncthreads();
    }

    // ---------------- layers 1..4 ----------------
    #pragma unroll
    for (int l = 1; l < LAYERS; ++l) {
      v8f acc[4];
      FragU A, B;
      loadA(A, &inbf[lr][0], hi * 8);           // ih kc=0, C = inline 0
      #pragma unroll
      for (int g = 0; g < 4; ++g) {
        loadBo(B, WihB, ((l - 1) * 60 + g * 15) * 1024);
        v8f z{};
        acc[g] = wmma_bf16(A, B, z);
      }
      #pragma unroll
      for (int kc = 1; kc < 3; ++kc) {          // ih kc=1..2
        loadA(A, &inbf[lr][0], kc * 32 + hi * 8);
        #pragma unroll
        for (int g = 0; g < 4; ++g) {
          loadBo(B, WihB, ((l - 1) * 60 + g * 15 + kc) * 1024);
          acc[g] = wmma_bf16(A, B, acc[g]);
        }
      }
      #pragma unroll
      for (int kc = 0; kc < 3; ++kc) {          // hh: LDS cache (l<4) / global (l==4)
        loadA(A, &hbf[l][lr][0], kc * 32 + hi * 8);
        #pragma unroll
        for (int g = 0; g < 4; ++g) {
          if (l < 4) loadBo(B, wChB, (l * 60 + g * 15 + kc) * 1024);
          else       loadBo(B, Whh4B, (g * 15 + kc) * 1024);
          acc[g] = wmma_bf16(A, B, acc[g]);
        }
      }
      // prefetch pipeline: next layer's Wih; at l==3 also Whh layer 4
      {
        const int nl = (l == LAYERS - 1) ? 1 : l + 1;   // l==4 -> next step's layer 1
        #pragma unroll
        for (int g = 0; g < 4; ++g)
          #pragma unroll
          for (int kc = 0; kc < 3; ++kc)
            prefetch_wgp((const char*)WihB + ((nl - 1) * 60 + g * 15 + kc) * 1024);
        if (l == 3) {
          #pragma unroll
          for (int g = 0; g < 4; ++g)
            #pragma unroll
            for (int kc = 0; kc < 3; ++kc)
              prefetch_wgp((const char*)Whh4B + (g * 15 + kc) * 1024);
        }
      }
      cellUpdate(l, acc[0], acc[1], acc[2], acc[3]);
      __syncthreads();
    }

    // ---------------- decoder: FC head + autoregressive feedback ----------
    if (!enc) {
      int s = t - SEQ;
      if (tid < BT) {
        float a = fcb0;
        for (int j = 0; j < HIDDEN; ++j) a += htop[tid][j] * sfcW[j];
        pred16[tid] = a;
        out[(size_t)(b0 + tid) * PRED_LEN + s] = a;
      }
      __syncthreads();
    }
  }
}

// ---------------------------------------------------------------------------
// ws layout (bytes):
//   eWih0 @0, eWih @20480, eWhh @266240, dWih0 @573440, dWih @593920,
//   dWhh @839680, eBias @1146880, dBias @1153280   (total 1159680 B)
// ---------------------------------------------------------------------------
extern "C" void kernel_launch(void* const* d_in, const int* in_sizes, int n_in,
                              void* d_out, int out_size, void* d_ws, size_t ws_size,
                              hipStream_t stream) {
  const float* x        = (const float*)d_in[0];
  const float* ff       = (const float*)d_in[1];
  const float* enc_Wih0 = (const float*)d_in[2];
  const float* enc_Wih  = (const float*)d_in[3];
  const float* enc_Whh  = (const float*)d_in[4];
  const float* enc_bih  = (const float*)d_in[5];
  const float* enc_bhh  = (const float*)d_in[6];
  const float* dec_Wih0 = (const float*)d_in[7];
  const float* dec_Wih  = (const float*)d_in[8];
  const float* dec_Whh  = (const float*)d_in[9];
  const float* dec_bih  = (const float*)d_in[10];
  const float* dec_bhh  = (const float*)d_in[11];
  const float* fc_W     = (const float*)d_in[12];
  const float* fc_b     = (const float*)d_in[13];
  float* out = (float*)d_out;

  unsigned short* wsu = (unsigned short*)d_ws;
  unsigned short* eWih0 = wsu + 0;
  unsigned short* eWih  = wsu + 10240;
  unsigned short* eWhh  = wsu + 133120;
  unsigned short* dWih0 = wsu + 286720;
  unsigned short* dWih  = wsu + 296960;
  unsigned short* dWhh  = wsu + 419840;
  float* eBias = (float*)((char*)d_ws + 1146880);
  float* dBias = (float*)((char*)d_ws + 1153280);

  conv_weights<<<20,  256, 0, stream>>>(enc_Wih0, INPUT_DIM, 1, eWih0);
  conv_weights<<<240, 256, 0, stream>>>(enc_Wih,  HIDDEN,    3, eWih);
  conv_weights<<<300, 256, 0, stream>>>(enc_Whh,  HIDDEN,    3, eWhh);
  conv_weights<<<20,  256, 0, stream>>>(dec_Wih0, INPUT_DIM, 1, dWih0);
  conv_weights<<<240, 256, 0, stream>>>(dec_Wih,  HIDDEN,    3, dWih);
  conv_weights<<<300, 256, 0, stream>>>(dec_Whh,  HIDDEN,    3, dWhh);
  conv_bias<<<7, 256, 0, stream>>>(enc_bih, enc_bhh, eBias, LAYERS * G);
  conv_bias<<<7, 256, 0, stream>>>(dec_bih, dec_bhh, dBias, LAYERS * G);

  // dynamic LDS: Wih0 + Whh layers 0..3 fragment cache (260 KB)
  lstm_seq2seq<<<BATCH / BT, NTHREADS, C_FRAGS * 1024, stream>>>(
      x, ff, eWih0, eWih, eWhh, eBias, dWih0, dWih, dWhh, dBias,
      fc_W, fc_b, out);
}
